// PointNetVAE_61546881352055
// MI455X (gfx1250) — matrix operand order repaired and checked by reading, not compiled, not run on hardware
//
#include <hip/hip_runtime.h>
#include <hip/hip_bf16.h>

// ---------------------------------------------------------------------------
// PointNet++ VAE forward for MI455X (gfx1250, wave32).
// - All dense math via v_wmma_f32_16x16x32_f16 (f16 in, f32 acc).
// - Weights pre-packed into WMMA B-fragment order (one v16h load per frag).
// - Activations kept in f16; GEMM X-tiles staged to LDS with
//   global_load_async_to_lds_b128 (+ s_wait_asynccnt), per cdna5 ISA ch.8.
// - SA modules fused (gather -> 3-layer MLP -> masked max) in LDS to avoid
//   ~1GB of HBM traffic for neighbor-expanded intermediates.
// ---------------------------------------------------------------------------

typedef __attribute__((ext_vector_type(16))) _Float16 v16h;
typedef __attribute__((ext_vector_type(8)))  float    v8f;

#define BN_SCALE 0.9999950000374997f   // 1/sqrt(1+1e-5)
#define NEG_BIG  (-1e30f)
#define CDIV(a,b) (((a)+(b)-1)/(b))

__device__ __forceinline__ v8f wmma16(v16h a, v16h b, v8f c) {
  return __builtin_amdgcn_wmma_f32_16x16x32_f16(
      false, a, false, b, (short)0, c, false, false);
}

// ISA 7.12.2 wave32 fragment maps.
// A 16x32 f16: lane L, elem e -> row=L&15, k = 16*(e>=8) + 8*(L>=16) + (e&7)
__device__ __forceinline__ int a_koff(int lane, int e) {
  return ((e >> 3) << 4) + ((lane >> 4) << 3) + (e & 7);
}

// Async copy of 16 bytes global -> LDS (ASYNCcnt-tracked).
__device__ __forceinline__ void async_cp16(unsigned lds_off, const void* gptr) {
  unsigned long long ga = (unsigned long long)(uintptr_t)gptr;
  asm volatile("global_load_async_to_lds_b128 %0, %1, off"
               :: "v"(lds_off), "v"(ga) : "memory");
}
__device__ __forceinline__ void async_wait0() {
  asm volatile("s_wait_asynccnt 0x0" ::: "memory");
}

// ---------------------------------------------------------------------------
// Weight packing: W[K][N] f32 -> f16 fragments, zero-padded to Kp x Np.
// out[((nt*ktiles+kb)*32 + lane)*16 + e] with k = kb*32 + 16*(e>=8) + 8*(lane>=16) + (e&7),
// n = nt*16 + (lane&15)   (B 32x16 fragment layout, ISA 7.12.2)
// ---------------------------------------------------------------------------
__global__ void pack_w_kernel(const float* __restrict__ W, int K, int N,
                              int ktiles, _Float16* __restrict__ out, int total) {
  int tid = blockIdx.x * blockDim.x + threadIdx.x;
  if (tid >= total) return;
  int e    = tid & 15;
  int lane = (tid >> 4) & 31;
  int tile = tid >> 9;
  int nt = tile / ktiles, kb = tile % ktiles;
  int k = kb * 32 + ((e >> 3) << 4) + ((lane >> 4) << 3) + (e & 7);
  int n = nt * 16 + (lane & 15);
  out[tid] = (k < K && n < N) ? (_Float16)W[(size_t)k * N + n] : (_Float16)0.0f;
}

__global__ void cvt_f16_kernel(const float* __restrict__ in,
                               _Float16* __restrict__ out, int total) {
  int tid = blockIdx.x * blockDim.x + threadIdx.x;
  if (tid >= total) return;
  out[tid] = (_Float16)in[tid];
}

// ---------------------------------------------------------------------------
// Farthest point sampling (matches jax.lax.scan: sel[0]=0, m-1 argmax steps).
// ---------------------------------------------------------------------------
__global__ void fps_kernel(const float* __restrict__ pos, int n, int m,
                           int* __restrict__ sel) {
  int b = blockIdx.x;
  pos += (size_t)b * n * 3;
  sel += (size_t)b * m;
  __shared__ float mind[2048];
  __shared__ float rv[256];
  __shared__ int   ri[256];
  __shared__ float cur[3];
  int t = threadIdx.x;
  if (t == 0) { cur[0] = pos[0]; cur[1] = pos[1]; cur[2] = pos[2]; sel[0] = 0; }
  __syncthreads();
  for (int j = t; j < n; j += blockDim.x) {
    float dx = pos[3*j] - cur[0], dy = pos[3*j+1] - cur[1], dz = pos[3*j+2] - cur[2];
    mind[j] = dx*dx + dy*dy + dz*dz;
  }
  __syncthreads();
  for (int it = 1; it < m; ++it) {
    float bv = -1.0f; int bi = n;
    for (int j = t; j < n; j += blockDim.x) {
      float v = mind[j];
      if (v > bv || (v == bv && j < bi)) { bv = v; bi = j; }
    }
    rv[t] = bv; ri[t] = bi;
    __syncthreads();
    for (int s = blockDim.x >> 1; s > 0; s >>= 1) {
      if (t < s && (rv[t+s] > rv[t] || (rv[t+s] == rv[t] && ri[t+s] < ri[t]))) {
        rv[t] = rv[t+s]; ri[t] = ri[t+s];
      }
      __syncthreads();
    }
    if (t == 0) {
      int i = ri[0]; sel[it] = i;
      cur[0] = pos[3*i]; cur[1] = pos[3*i+1]; cur[2] = pos[3*i+2];
    }
    __syncthreads();
    for (int j = t; j < n; j += blockDim.x) {
      float dx = pos[3*j] - cur[0], dy = pos[3*j+1] - cur[1], dz = pos[3*j+2] - cur[2];
      mind[j] = fminf(mind[j], dx*dx + dy*dy + dz*dz);
    }
    __syncthreads();
  }
}

__global__ void gather_pos_kernel(const float* __restrict__ pos, int n,
                                  const int* __restrict__ sel, int m,
                                  float* __restrict__ qpos, int total) {
  int tid = blockIdx.x * blockDim.x + threadIdx.x;
  if (tid >= total) return;
  int b = tid / m;
  int s = sel[tid];
  const float* pp = pos + ((size_t)b * n + s) * 3;
  float* o = qpos + (size_t)tid * 3;
  o[0] = pp[0]; o[1] = pp[1]; o[2] = pp[2];
}

// First K=64 in-radius indices in scan order (== K smallest indices), padded.
__global__ void radius_kernel(const float* __restrict__ qpos,
                              const float* __restrict__ pos, int n, int m,
                              float r2, int* __restrict__ idx,
                              int* __restrict__ valid, int total) {
  int tid = blockIdx.x * blockDim.x + threadIdx.x;
  if (tid >= total) return;
  int b = tid / m;
  const float* qp = qpos + (size_t)tid * 3;
  const float* pp = pos + (size_t)b * n * 3;
  int* oi = idx + (size_t)tid * 64;
  int* ov = valid + (size_t)tid * 64;
  int cnt = 0;
  for (int j = 0; j < n && cnt < 64; ++j) {
    float dx = pp[3*j] - qp[0], dy = pp[3*j+1] - qp[1], dz = pp[3*j+2] - qp[2];
    float d = dx*dx + dy*dy + dz*dz;
    if (d <= r2) { oi[cnt] = j; ov[cnt] = 1; ++cnt; }
  }
  for (; cnt < 64; ++cnt) { oi[cnt] = n - 1; ov[cnt] = 0; }
}

__global__ void knn_kernel(const float* __restrict__ qpos,
                           const float* __restrict__ pos, int n, int m,
                           int* __restrict__ nbr, int total) {
  int tid = blockIdx.x * blockDim.x + threadIdx.x;
  if (tid >= total) return;
  int b = tid / m;
  const float* qp = qpos + (size_t)tid * 3;
  const float* pp = pos + (size_t)b * n * 3;
  float bd[16]; int bi[16];
  #pragma unroll
  for (int i = 0; i < 16; ++i) { bd[i] = 3.4e38f; bi[i] = 0; }
  for (int j = 0; j < n; ++j) {
    float dx = pp[3*j] - qp[0], dy = pp[3*j+1] - qp[1], dz = pp[3*j+2] - qp[2];
    float d = dx*dx + dy*dy + dz*dz;
    if (d < bd[15]) {
      int p = 15;
      while (p > 0 && bd[p-1] > d) { bd[p] = bd[p-1]; bi[p] = bi[p-1]; --p; }
      bd[p] = d; bi[p] = j;
    }
  }
  int* o = nbr + (size_t)tid * 16;
  #pragma unroll
  for (int i = 0; i < 16; ++i) o[i] = bi[i];
}

// ---------------------------------------------------------------------------
// One MLP layer over 64 LDS rows, packed weights; wave = 16 rows.
// ---------------------------------------------------------------------------
__device__ __attribute__((always_inline)) inline
void layer_ff(const _Float16* in, int Kp, const v16h* Wp, int ktiles, int Nw,
              const float* bias, const float* g, const float* bt,
              _Float16* out, int lane, int w) {
  for (int nt = 0; nt < Nw / 16; ++nt) {
    v8f c = {};
    int nn = nt * 16 + (lane & 15);
    for (int kb = 0; kb < Kp; kb += 32) {
      v16h a;
      #pragma unroll
      for (int e = 0; e < 16; ++e)
        a[e] = in[(w * 16 + (lane & 15)) * Kp + kb + a_koff(lane, e)];
      v16h bf = Wp[((size_t)nt * ktiles + (kb >> 5)) * 32 + lane];
      c = wmma16(a, bf, c);
    }
    float bb = bias[nn];
    float gg = g ? g[nn] : 0.0f;
    float tt = bt ? bt[nn] : 0.0f;
    #pragma unroll
    for (int r = 0; r < 8; ++r) {
      int mm = w * 16 + r + ((lane >> 4) << 3);
      float v = c[r] + bb;
      if (g) { v = gg * (v * BN_SCALE) + tt; v = fmaxf(v, 0.0f); }
      out[mm * Nw + nn] = (_Float16)v;
    }
  }
}

// ---------------------------------------------------------------------------
// Fused SA module: block (4 waves) per query: gather 64 neighbor rows of
// [x_j, pos_j - pos_i] -> 3-layer WMMA MLP -> masked max over rows.
// ---------------------------------------------------------------------------
__global__ void __launch_bounds__(128)
sa_fused_kernel(const _Float16* __restrict__ xin, int n, int Cx,
                const float* __restrict__ pos,
                const float* __restrict__ qpos, int m,
                const int* __restrict__ idx, const int* __restrict__ valid,
                const _Float16* __restrict__ Wp0, const float* __restrict__ b0,
                const float* __restrict__ g0, const float* __restrict__ bt0, int N0,
                const _Float16* __restrict__ Wp1, const float* __restrict__ b1,
                const float* __restrict__ g1, const float* __restrict__ bt1, int N1,
                const _Float16* __restrict__ Wp2, const float* __restrict__ b2, int N2,
                _Float16* __restrict__ xout) {
  __shared__ _Float16 bufA[64 * 288];   // gather / layer1-out (Kp <= 288)
  __shared__ _Float16 bufB[64 * 128];   // layer0-out          (N0 <= 128)
  __shared__ int   src[64];
  __shared__ int   vmask[64];
  __shared__ float qp[3];
  __shared__ float red[64];

  int b = blockIdx.y, q = blockIdx.x;
  int t = threadIdx.x, lane = t & 31, w = t >> 5;
  size_t bq = (size_t)b * m + q;

  if (t < 64) {
    src[t]   = idx[bq * 64 + t];
    vmask[t] = valid[bq * 64 + t];
  }
  if (t < 3) qp[t] = qpos[bq * 3 + t];
  __syncthreads();

  int K0 = Cx + 3;
  int K0p = (K0 + 31) & ~31;

  // gather (three unguarded loops: copy x, pos-diff, zero pad)
  for (int e = t; e < 64 * Cx; e += 128) {
    int row = e / Cx, c = e % Cx;
    bufA[row * K0p + c] = xin[((size_t)b * n + src[row]) * Cx + c];
  }
  for (int e = t; e < 64 * 3; e += 128) {
    int row = e / 3, d = e % 3;
    bufA[row * K0p + Cx + d] =
        (_Float16)(pos[((size_t)b * n + src[row]) * 3 + d] - qp[d]);
  }
  int pad = K0p - K0;
  for (int e = t; e < 64 * pad; e += 128) {
    int row = e / pad, c = K0 + e % pad;
    bufA[row * K0p + c] = (_Float16)0.0f;
  }
  __syncthreads();

  layer_ff(bufA, K0p, (const v16h*)Wp0, K0p >> 5, N0, b0, g0, bt0, bufB, lane, w);
  __syncthreads();
  layer_ff(bufB, N0, (const v16h*)Wp1, N0 >> 5, N1, b1, g1, bt1, bufA, lane, w);
  __syncthreads();

  // last layer + masked scatter-max over the 64 rows
  const v16h* Wp2v = (const v16h*)Wp2;
  int kt2 = N1 >> 5;
  for (int nt = 0; nt < N2 / 16; ++nt) {
    v8f c = {};
    int nn = nt * 16 + (lane & 15);
    for (int kb = 0; kb < N1; kb += 32) {
      v16h a;
      #pragma unroll
      for (int e = 0; e < 16; ++e)
        a[e] = bufA[(w * 16 + (lane & 15)) * N1 + kb + a_koff(lane, e)];
      v16h bf = Wp2v[((size_t)nt * kt2 + (kb >> 5)) * 32 + lane];
      c = wmma16(a, bf, c);
    }
    float bb = b2[nn];
    float part = NEG_BIG;
    #pragma unroll
    for (int r = 0; r < 8; ++r) {
      int mm = w * 16 + r + ((lane >> 4) << 3);
      float v = c[r] + bb;
      part = vmask[mm] ? fmaxf(part, v) : part;
    }
    part = fmaxf(part, __shfl_xor(part, 16, 32));
    if (lane < 16) red[w * 16 + lane] = part;
    __syncthreads();
    if (t < 16) {
      float v = fmaxf(fmaxf(red[t], red[t + 16]), fmaxf(red[t + 32], red[t + 48]));
      xout[bq * N2 + nt * 16 + t] = (_Float16)v;
    }
    __syncthreads();
  }
}

// ---------------------------------------------------------------------------
// Generic GEMM, f16 activations + packed f16 weights.
// Block = 128 threads (4 waves), tile 64(M) x 16(N); X tile (64x32 f16 = 4KB)
// staged via global_load_async_to_lds_b128. Kp must be a multiple of 32 and
// the X buffer padded to 64-row multiples (done at allocation).
// ---------------------------------------------------------------------------
template <bool F32OUT>
__global__ void __launch_bounds__(128)
gemm_kernel(const _Float16* __restrict__ X, int M, int Kp,
            const _Float16* __restrict__ Wpack, int N,
            const float* __restrict__ bias,
            const float* __restrict__ g, const float* __restrict__ bt,
            void* __restrict__ Yv) {
  __shared__ _Float16 xs[64 * 32];
  int t = threadIdx.x, lane = t & 31, w = t >> 5;
  int mBase = blockIdx.x * 64;
  int ktiles = Kp >> 5;
  const v16h* Wp = (const v16h*)Wpack;
  int nn = blockIdx.y * 16 + (lane & 15);
  unsigned ldsBase = (unsigned)(uintptr_t)(&xs[0]);
  v8f c = {};
  for (int kb = 0; kb < Kp; kb += 32) {
    // async-stage X tile: 64 rows x 64B; 256 x 16B chunks over 128 threads
    #pragma unroll
    for (int i = 0; i < 2; ++i) {
      int ob   = (t + 128 * i) * 16;          // byte offset in tile
      int row  = ob >> 6;
      int colb = ob & 63;
      const char* gp = (const char*)(X + (size_t)(mBase + row) * Kp + kb) + colb;
      async_cp16(ldsBase + (unsigned)ob, gp);
    }
    if (kb + 32 < Kp)
      __builtin_prefetch(&Wp[((size_t)blockIdx.y * ktiles + (kb >> 5) + 1) * 32 + lane], 0, 1);
    async_wait0();
    __syncthreads();
    v16h a;
    #pragma unroll
    for (int e = 0; e < 16; ++e)
      a[e] = xs[(w * 16 + (lane & 15)) * 32 + a_koff(lane, e)];
    v16h bf = Wp[((size_t)blockIdx.y * ktiles + (kb >> 5)) * 32 + lane];
    c = wmma16(a, bf, c);
    __syncthreads();
  }
  if (nn < N) {
    float bb = bias[nn];
    float gg = g ? g[nn] : 0.0f;
    float tt = bt ? bt[nn] : 0.0f;
    #pragma unroll
    for (int r = 0; r < 8; ++r) {
      int mm = mBase + w * 16 + r + ((lane >> 4) << 3);
      if (mm < M) {
        float v = c[r] + bb;
        if (g) { v = gg * (v * BN_SCALE) + tt; v = fmaxf(v, 0.0f); }
        if (F32OUT) ((float*)Yv)[(size_t)mm * N + nn] = v;
        else        ((_Float16*)Yv)[(size_t)mm * N + nn] = (_Float16)v;
      }
    }
  }
}

__global__ void gather_max_kernel(const _Float16* __restrict__ h, int n, int F,
                                  const int* __restrict__ nbr, int m,
                                  _Float16* __restrict__ out, int total) {
  int tid = blockIdx.x * blockDim.x + threadIdx.x;
  if (tid >= total) return;
  int f = tid % F; int bq = tid / F; int b = bq / m;
  const int* nb = nbr + (size_t)bq * 16;
  float v = NEG_BIG;
  #pragma unroll 4
  for (int j = 0; j < 16; ++j)
    v = fmaxf(v, (float)h[((size_t)b * n + nb[j]) * F + f]);
  out[(size_t)bq * F + f] = (_Float16)v;
}

__global__ void mean_pool_kernel(const _Float16* __restrict__ h, int rows, int F,
                                 _Float16* __restrict__ out, int total) {
  int tid = blockIdx.x * blockDim.x + threadIdx.x;
  if (tid >= total) return;
  int b = tid / F, f = tid % F;
  float s = 0.0f;
  for (int i = 0; i < rows; ++i) s += (float)h[((size_t)b * rows + i) * F + f];
  out[tid] = (_Float16)(s / (float)rows);
}

// cat[row][0:1024]=x2d, [1024:1027]=p2d, [1027:1056]=0  (row stride 1056)
__global__ void concat_kernel(const _Float16* __restrict__ x,
                              const float* __restrict__ p,
                              _Float16* __restrict__ out, int total) {
  int tid = blockIdx.x * blockDim.x + threadIdx.x;
  if (tid >= total) return;
  int row = tid / 1056, f = tid % 1056;
  _Float16 v = (_Float16)0.0f;
  if (f < 1024)      v = x[(size_t)row * 1024 + f];
  else if (f < 1027) v = (_Float16)p[(size_t)row * 3 + (f - 1024)];
  out[tid] = v;
}

__global__ void latent_kernel(const _Float16* __restrict__ mu,
                              const _Float16* __restrict__ lv,
                              const float* __restrict__ eps,
                              _Float16* __restrict__ z, int total) {
  int tid = blockIdx.x * blockDim.x + threadIdx.x;
  if (tid >= total) return;
  z[tid] = (_Float16)((float)mu[tid] + eps[tid] * expf(0.5f * (float)lv[tid]));
}

// ---------------------------------------------------------------------------
static void run_pack(const float* W, int K, int N, _Float16* out, hipStream_t s) {
  int kt = CDIV(K, 32), ntl = CDIV(N, 16);
  int total = kt * ntl * 512;
  pack_w_kernel<<<dim3((unsigned)CDIV(total, 256)), dim3(256), 0, s>>>(W, K, N, kt, out, total);
}
static void run_gemm16(const _Float16* X, int M, int Kp, const _Float16* Wp, int N,
                       const float* b, const float* g, const float* bt,
                       _Float16* Y, hipStream_t s) {
  dim3 grid((unsigned)CDIV(M, 64), (unsigned)CDIV(N, 16));
  gemm_kernel<false><<<grid, dim3(128), 0, s>>>(X, M, Kp, Wp, N, b, g, bt, (void*)Y);
}
static void run_gemm32(const _Float16* X, int M, int Kp, const _Float16* Wp, int N,
                       const float* b, float* Y, hipStream_t s) {
  dim3 grid((unsigned)CDIV(M, 64), (unsigned)CDIV(N, 16));
  gemm_kernel<true><<<grid, dim3(128), 0, s>>>(X, M, Kp, Wp, N, b, nullptr, nullptr, (void*)Y);
}

extern "C" void kernel_launch(void* const* d_in, const int* in_sizes, int n_in,
                              void* d_out, int out_size, void* d_ws, size_t ws_size,
                              hipStream_t stream) {
  (void)in_sizes; (void)n_in; (void)out_size; (void)ws_size;
  const int B = 16, N = 2048;
  #define FP(i) ((const float*)d_in[(i)])
  const float* data = FP(0);
  const float* eps  = FP(1);
  const float *sa1W0=FP(2),*sa1b0=FP(3),*sa1g0=FP(4),*sa1t0=FP(5);
  const float *sa1W1=FP(6),*sa1b1=FP(7),*sa1g1=FP(8),*sa1t1=FP(9);
  const float *sa1W2=FP(10),*sa1b2=FP(11);
  const float *td1W0=FP(12),*td1b0=FP(13),*td1g0=FP(14),*td1t0=FP(15);
  const float *td1W1=FP(16),*td1b1=FP(17);
  const float *sa2W0=FP(18),*sa2b0=FP(19),*sa2g0=FP(20),*sa2t0=FP(21);
  const float *sa2W1=FP(22),*sa2b1=FP(23),*sa2g1=FP(24),*sa2t1=FP(25);
  const float *sa2W2=FP(26),*sa2b2=FP(27);
  const float *td2W0=FP(28),*td2b0=FP(29),*td2g0=FP(30),*td2t0=FP(31);
  const float *td2W1=FP(32),*td2b1=FP(33);
  const float *sa3W0=FP(34),*sa3b0=FP(35),*sa3g0=FP(36),*sa3t0=FP(37);
  const float *sa3W1=FP(38),*sa3b1=FP(39),*sa3g1=FP(40),*sa3t1=FP(41);
  const float *sa3W2=FP(42),*sa3b2=FP(43);
  const float *e1W0=FP(44),*e1b0=FP(45),*e1g0=FP(46),*e1t0=FP(47);
  const float *e1W1=FP(48),*e1b1=FP(49);
  const float *emW=FP(50),*emb=FP(51);
  const float *evW=FP(52),*evb=FP(53);
  const float *dcW0=FP(54),*dcb0=FP(55),*dcg0=FP(56),*dct0=FP(57);
  const float *dcW1=FP(58),*dcb1=FP(59);
  const float *d2W=FP(60),*d2b=FP(61);
  const float *hW0=FP(62),*hb0=FP(63),*hg0=FP(64),*ht0=FP(65);
  const float *hW1=FP(66),*hb1=FP(67),*hg1=FP(68),*ht1=FP(69);
  const float *hW2=FP(70),*hb2=FP(71);

  char* wsp = (char*)d_ws; size_t cur = 0;
  auto alloc = [&](size_t bytes) -> char* {
    char* p = wsp + cur; cur = (cur + bytes + 255) & ~(size_t)255; return p;
  };
  #define PACKB(K,Nn) ((size_t)CDIV(K,32) * CDIV(Nn,16) * 512 * 2)
  // packed weights (f16)
  _Float16* Psa1W0=(_Float16*)alloc(PACKB(6,64));
  _Float16* Psa1W1=(_Float16*)alloc(PACKB(64,64));
  _Float16* Psa1W2=(_Float16*)alloc(PACKB(64,128));
  _Float16* Ptd1W0=(_Float16*)alloc(PACKB(128,256));
  _Float16* Ptd1W1=(_Float16*)alloc(PACKB(256,256));
  _Float16* Psa2W0=(_Float16*)alloc(PACKB(259,128));
  _Float16* Psa2W1=(_Float16*)alloc(PACKB(128,128));
  _Float16* Psa2W2=(_Float16*)alloc(PACKB(128,512));
  _Float16* Ptd2W0=(_Float16*)alloc(PACKB(512,1024));
  _Float16* Ptd2W1=(_Float16*)alloc(PACKB(1024,1024));
  _Float16* Psa3W0=(_Float16*)alloc(PACKB(1027,512));
  _Float16* Psa3W1=(_Float16*)alloc(PACKB(512,512));
  _Float16* Psa3W2=(_Float16*)alloc(PACKB(512,1024));
  _Float16* Pe1W0 =(_Float16*)alloc(PACKB(1024,512));
  _Float16* Pe1W1 =(_Float16*)alloc(PACKB(512,256));
  _Float16* PemW  =(_Float16*)alloc(PACKB(256,128));
  _Float16* PevW  =(_Float16*)alloc(PACKB(256,128));
  _Float16* PdcW0 =(_Float16*)alloc(PACKB(128,256));
  _Float16* PdcW1 =(_Float16*)alloc(PACKB(256,512));
  _Float16* Pd2W  =(_Float16*)alloc(PACKB(512,1024));
  _Float16* PhW0  =(_Float16*)alloc(PACKB(1024,512));
  _Float16* PhW1  =(_Float16*)alloc(PACKB(512,256));
  _Float16* PhW2  =(_Float16*)alloc(PACKB(256,40));
  // activations / graph state
  _Float16* datah=(_Float16*)alloc((size_t)B*2048*3*2);
  int*   sel1  =(int*)  alloc((size_t)B*1024*4);
  float* qpos1 =(float*)alloc((size_t)B*1024*3*4);
  int*   idx1  =(int*)  alloc((size_t)B*1024*64*4);
  int*   val1  =(int*)  alloc((size_t)B*1024*64*4);
  _Float16* x1 =(_Float16*)alloc((size_t)B*1024*128*2);
  int*   selA  =(int*)  alloc((size_t)B*256*4);
  float* p1d   =(float*)alloc((size_t)B*256*3*4);
  int*   nbr1  =(int*)  alloc((size_t)B*256*16*4);
  _Float16* h1a=(_Float16*)alloc((size_t)B*1024*256*2);
  _Float16* h1b=(_Float16*)alloc((size_t)B*1024*256*2);
  _Float16* x1d=(_Float16*)alloc((size_t)B*256*256*2);
  int*   sel2  =(int*)  alloc((size_t)B*64*4);
  float* qpos2 =(float*)alloc((size_t)B*64*3*4);
  int*   idx2  =(int*)  alloc((size_t)B*64*64*4);
  int*   val2  =(int*)  alloc((size_t)B*64*64*4);
  _Float16* x2 =(_Float16*)alloc((size_t)B*64*512*2);
  int*   selB  =(int*)  alloc((size_t)B*16*4);
  float* p2d   =(float*)alloc((size_t)B*16*3*4);
  int*   nbr2  =(int*)  alloc((size_t)B*16*16*4);
  _Float16* h2a=(_Float16*)alloc((size_t)B*64*1024*2);
  _Float16* h2b=(_Float16*)alloc((size_t)B*64*1024*2);
  _Float16* x2d=(_Float16*)alloc((size_t)B*16*1024*2);
  _Float16* cat3=(_Float16*)alloc((size_t)B*16*1056*2);
  _Float16* h3a=(_Float16*)alloc((size_t)B*16*512*2);
  _Float16* h3b=(_Float16*)alloc((size_t)B*16*512*2);
  _Float16* h3c=(_Float16*)alloc((size_t)B*16*1024*2);
  // tail buffers padded to 64 rows (async staging reads full 64-row tiles)
  _Float16* gbuf=(_Float16*)alloc((size_t)64*1024*2);
  _Float16* e1a =(_Float16*)alloc((size_t)64*512*2);
  _Float16* e1b =(_Float16*)alloc((size_t)64*256*2);
  _Float16* mu  =(_Float16*)alloc((size_t)64*128*2);
  _Float16* lv  =(_Float16*)alloc((size_t)64*128*2);
  _Float16* zb  =(_Float16*)alloc((size_t)64*128*2);
  _Float16* d1a =(_Float16*)alloc((size_t)64*256*2);
  _Float16* d1b =(_Float16*)alloc((size_t)64*512*2);
  _Float16* decb=(_Float16*)alloc((size_t)64*1024*2);
  _Float16* t1  =(_Float16*)alloc((size_t)64*512*2);
  _Float16* t2  =(_Float16*)alloc((size_t)64*256*2);

  #define L1D(kern, total, ...) \
    kern<<<dim3((unsigned)CDIV((total),256)), dim3(256), 0, stream>>>(__VA_ARGS__)

  // ---- pack all weights into WMMA fragment order --------------------------
  run_pack(sa1W0,6,64,Psa1W0,stream);   run_pack(sa1W1,64,64,Psa1W1,stream);
  run_pack(sa1W2,64,128,Psa1W2,stream);
  run_pack(td1W0,128,256,Ptd1W0,stream);run_pack(td1W1,256,256,Ptd1W1,stream);
  run_pack(sa2W0,259,128,Psa2W0,stream);run_pack(sa2W1,128,128,Psa2W1,stream);
  run_pack(sa2W2,128,512,Psa2W2,stream);
  run_pack(td2W0,512,1024,Ptd2W0,stream);run_pack(td2W1,1024,1024,Ptd2W1,stream);
  run_pack(sa3W0,1027,512,Psa3W0,stream);run_pack(sa3W1,512,512,Psa3W1,stream);
  run_pack(sa3W2,512,1024,Psa3W2,stream);
  run_pack(e1W0,1024,512,Pe1W0,stream); run_pack(e1W1,512,256,Pe1W1,stream);
  run_pack(emW,256,128,PemW,stream);    run_pack(evW,256,128,PevW,stream);
  run_pack(dcW0,128,256,PdcW0,stream);  run_pack(dcW1,256,512,PdcW1,stream);
  run_pack(d2W,512,1024,Pd2W,stream);
  run_pack(hW0,1024,512,PhW0,stream);   run_pack(hW1,512,256,PhW1,stream);
  run_pack(hW2,256,40,PhW2,stream);
  L1D(cvt_f16_kernel, B*2048*3, data, datah, B*2048*3);

  // ---- SA1 ----------------------------------------------------------------
  fps_kernel<<<dim3(B), dim3(256), 0, stream>>>(data, N, 1024, sel1);
  L1D(gather_pos_kernel, B*1024, data, N, sel1, 1024, qpos1, B*1024);
  L1D(radius_kernel, B*1024, qpos1, data, N, 1024, 0.04f, idx1, val1, B*1024);
  sa_fused_kernel<<<dim3(1024, B), dim3(128), 0, stream>>>(
      datah, N, 3, data, qpos1, 1024, idx1, val1,
      Psa1W0, sa1b0, sa1g0, sa1t0, 64,
      Psa1W1, sa1b1, sa1g1, sa1t1, 64,
      Psa1W2, sa1b2, 128, x1);

  // ---- TD1 ----------------------------------------------------------------
  fps_kernel<<<dim3(B), dim3(256), 0, stream>>>(qpos1, 1024, 256, selA);
  L1D(gather_pos_kernel, B*256, qpos1, 1024, selA, 256, p1d, B*256);
  L1D(knn_kernel, B*256, p1d, qpos1, 1024, 256, nbr1, B*256);
  run_gemm16(x1, B*1024, 128, Ptd1W0, 256, td1b0, td1g0, td1t0, h1a, stream);
  run_gemm16(h1a, B*1024, 256, Ptd1W1, 256, td1b1, nullptr, nullptr, h1b, stream);
  L1D(gather_max_kernel, B*256*256, h1b, 1024, 256, nbr1, 256, x1d, B*256*256);

  // ---- SA2 ----------------------------------------------------------------
  fps_kernel<<<dim3(B), dim3(256), 0, stream>>>(p1d, 256, 64, sel2);
  L1D(gather_pos_kernel, B*64, p1d, 256, sel2, 64, qpos2, B*64);
  L1D(radius_kernel, B*64, qpos2, p1d, 256, 64, 0.16f, idx2, val2, B*64);
  sa_fused_kernel<<<dim3(64, B), dim3(128), 0, stream>>>(
      x1d, 256, 256, p1d, qpos2, 64, idx2, val2,
      Psa2W0, sa2b0, sa2g0, sa2t0, 128,
      Psa2W1, sa2b1, sa2g1, sa2t1, 128,
      Psa2W2, sa2b2, 512, x2);

  // ---- TD2 ----------------------------------------------------------------
  fps_kernel<<<dim3(B), dim3(256), 0, stream>>>(qpos2, 64, 16, selB);
  L1D(gather_pos_kernel, B*16, qpos2, 64, selB, 16, p2d, B*16);
  L1D(knn_kernel, B*16, p2d, qpos2, 64, 16, nbr2, B*16);
  run_gemm16(x2, B*64, 512, Ptd2W0, 1024, td2b0, td2g0, td2t0, h2a, stream);
  run_gemm16(h2a, B*64, 1024, Ptd2W1, 1024, td2b1, nullptr, nullptr, h2b, stream);
  L1D(gather_max_kernel, B*16*1024, h2b, 64, 1024, nbr2, 16, x2d, B*16*1024);

  // ---- SA3 + pooling ------------------------------------------------------
  L1D(concat_kernel, B*16*1056, x2d, p2d, cat3, B*16*1056);
  run_gemm16(cat3, B*16, 1056, Psa3W0, 512, sa3b0, sa3g0, sa3t0, h3a, stream);
  run_gemm16(h3a, B*16, 512, Psa3W1, 512, sa3b1, sa3g1, sa3t1, h3b, stream);
  run_gemm16(h3b, B*16, 512, Psa3W2, 1024, sa3b2, nullptr, nullptr, h3c, stream);
  L1D(mean_pool_kernel, B*1024, h3c, 16, 1024, gbuf, B*1024);

  // ---- VAE encoder / reparam / decoder / head -----------------------------
  run_gemm16(gbuf, B, 1024, Pe1W0, 512, e1b0, e1g0, e1t0, e1a, stream);
  run_gemm16(e1a, B, 512, Pe1W1, 256, e1b1, nullptr, nullptr, e1b, stream);
  run_gemm16(e1b, B, 256, PemW, 128, emb, nullptr, nullptr, mu, stream);
  run_gemm16(e1b, B, 256, PevW, 128, evb, nullptr, nullptr, lv, stream);
  L1D(latent_kernel, B*128, mu, lv, eps, zb, B*128);
  run_gemm16(zb, B, 128, PdcW0, 256, dcb0, dcg0, dct0, d1a, stream);
  run_gemm16(d1a, B, 256, PdcW1, 512, dcb1, nullptr, nullptr, d1b, stream);
  run_gemm16(d1b, B, 512, Pd2W, 1024, d2b, nullptr, nullptr, decb, stream);
  run_gemm16(decb, B, 1024, PhW0, 512, hb0, hg0, ht0, t1, stream);
  run_gemm16(t1, B, 512, PhW1, 256, hb1, hg1, ht1, t2, stream);
  run_gemm32(t2, B, 256, PhW2, 40, hb2, (float*)d_out, stream);
}
